// UTT_71811853189765
// MI455X (gfx1250) — matrix operand order repaired and checked by reading, not compiled, not run on hardware
//
#include <hip/hip_runtime.h>
#include <hip/hip_bf16.h>

// ---------------------------------------------------------------------------
// Model constants
// ---------------------------------------------------------------------------
#define HID   512
#define HEADS 8
#define DH    64
#define NLAY  4
#define FFN   2048
#define LSQ   256      // sequence length L
#define BS    32
#define TTK   64
#define PPN   128
#define KC    10
#define RFD   32
#define POUT  64
#define NEGV  (-1e9f)
#define INDIM 1090     // 2*(HID+RFD+1)
#define XPAD  1092     // INDIM padded to multiple of 4 for vector loads

typedef __attribute__((ext_vector_type(16))) __bf16 bf16x16;
typedef __attribute__((ext_vector_type(8)))  __bf16 bf16x8;
typedef __attribute__((ext_vector_type(2)))  __bf16 bf16x2;
typedef __attribute__((ext_vector_type(8)))  float  f32x8;

// ---------------------------------------------------------------------------
// Generic batched WMMA GEMM:  C[z] = A[z] @ B[z] (+bias) (+epilogue)
//   A: row-major [M,K] (lda),   B: row-major [K,N] (ldb) or transB [N,K]
//   batch z decomposed as (z/heads, z%heads) with two strides per operand.
//   epi: 0=none 1=relu 2=tanh 3=attention mask+scale (uses mlen, scale)
// Block: 128 threads = 4 waves (2x2); block tile 64(M) x 128(N); K-step 32.
// Each wave computes a 32x64 sub-tile = 2x4 WMMA 16x16x32 ops.
// ---------------------------------------------------------------------------
template<bool TRANSB>
__global__ __launch_bounds__(128)
void gemm_wmma(const float* __restrict__ A, const float* __restrict__ B,
               const float* __restrict__ bias, float* __restrict__ C,
               int M, int N, int K, int lda, int ldb, int ldc,
               int heads,
               long long sAo, long long sAi,
               long long sBo, long long sBi,
               long long sCo, long long sCi,
               int epi, float scale, const int* __restrict__ mlen)
{
  // rows padded to 40 elements (80B = multiple of 16B) for aligned b128 LDS ops
  __shared__ __align__(16) __bf16 As[64][40];
  __shared__ __align__(16) __bf16 Bs[128][40];

  const int tid  = threadIdx.x;
  const int lane = tid & 31;
  const int wv   = tid >> 5;
  const int wm   = wv & 1;        // wave row (2x2 wave grid)
  const int wn   = wv >> 1;       // wave col
  const int z    = blockIdx.z;
  const int zo   = z / heads, zi = z - zo * heads;
  const long long offA = (long long)zo * sAo + (long long)zi * sAi;
  const long long offB = (long long)zo * sBo + (long long)zi * sBi;
  const long long offC = (long long)zo * sCo + (long long)zi * sCi;
  const int m0 = blockIdx.y * 64;
  const int n0 = blockIdx.x * 128;

  // block-uniform vector-path predicates (alignment only; rows are guarded
  // per-thread with a single divergent branch around the loads)
  const bool vecA = ((lda & 3) == 0);
  const bool vecB = TRANSB ? ((ldb & 3) == 0) : true;

  f32x8 acc[2][4] = {};

  const int rsel  = lane & 15;
  const int khalf = lane >> 4;

  // per-thread coordinates for the A vector loader (16 elems / thread)
  const int lr  = tid >> 1;          // row 0..63
  const int lc0 = (tid & 1) * 16;    // col 0 or 16

  for (int k0 = 0; k0 < K; k0 += 32) {
    const bool fullK = (k0 + 32 <= K);
    // ================= load A tile 64x32 (f32 -> bf16) =================
    if (vecA && fullK) {
      int gr = m0 + lr;
      bf16x8 h0 = {}, h1 = {};
      if (gr < M) {                       // one divergent branch per thread
        const float* gp = A + offA + (long long)gr * lda + (k0 + lc0);
        float4 f0 = ((const float4*)gp)[0];
        float4 f1 = ((const float4*)gp)[1];
        float4 f2 = ((const float4*)gp)[2];
        float4 f3 = ((const float4*)gp)[3];
        if (k0 + 32 < K) __builtin_prefetch(gp + 32, 0, 3);   // next K tile
        h0[0]=(__bf16)f0.x; h0[1]=(__bf16)f0.y; h0[2]=(__bf16)f0.z; h0[3]=(__bf16)f0.w;
        h0[4]=(__bf16)f1.x; h0[5]=(__bf16)f1.y; h0[6]=(__bf16)f1.z; h0[7]=(__bf16)f1.w;
        h1[0]=(__bf16)f2.x; h1[1]=(__bf16)f2.y; h1[2]=(__bf16)f2.z; h1[3]=(__bf16)f2.w;
        h1[4]=(__bf16)f3.x; h1[5]=(__bf16)f3.y; h1[6]=(__bf16)f3.z; h1[7]=(__bf16)f3.w;
      }
      *(bf16x8*)(&As[lr][lc0])     = h0;
      *(bf16x8*)(&As[lr][lc0 + 8]) = h1;
    } else {
      for (int i = tid; i < 64 * 32; i += 128) {   // rolled: edge tiles only
        int r = i >> 5, c = i & 31;
        int gr = m0 + r, gc = k0 + c;
        float v = 0.0f;
        if (gr < M && gc < K) v = A[offA + (long long)gr * lda + gc];
        As[r][c] = (__bf16)v;
      }
    }
    // ================= load B tile as Bs[n][k], n-tile = 128 =================
    if (TRANSB) {
      if (vecB && fullK) {
        int gn = n0 + tid;                 // one row (32 K-vals) per thread
        bf16x8 h0 = {}, h1 = {}, h2 = {}, h3 = {};
        if (gn < N) {
          const float* gp = B + offB + (long long)gn * ldb + k0;
          float4 f0 = ((const float4*)gp)[0];
          float4 f1 = ((const float4*)gp)[1];
          float4 f2 = ((const float4*)gp)[2];
          float4 f3 = ((const float4*)gp)[3];
          float4 f4 = ((const float4*)gp)[4];
          float4 f5 = ((const float4*)gp)[5];
          float4 f6 = ((const float4*)gp)[6];
          float4 f7 = ((const float4*)gp)[7];
          if (k0 + 32 < K) __builtin_prefetch(gp + 32, 0, 3);
          h0[0]=(__bf16)f0.x; h0[1]=(__bf16)f0.y; h0[2]=(__bf16)f0.z; h0[3]=(__bf16)f0.w;
          h0[4]=(__bf16)f1.x; h0[5]=(__bf16)f1.y; h0[6]=(__bf16)f1.z; h0[7]=(__bf16)f1.w;
          h1[0]=(__bf16)f2.x; h1[1]=(__bf16)f2.y; h1[2]=(__bf16)f2.z; h1[3]=(__bf16)f2.w;
          h1[4]=(__bf16)f3.x; h1[5]=(__bf16)f3.y; h1[6]=(__bf16)f3.z; h1[7]=(__bf16)f3.w;
          h2[0]=(__bf16)f4.x; h2[1]=(__bf16)f4.y; h2[2]=(__bf16)f4.z; h2[3]=(__bf16)f4.w;
          h2[4]=(__bf16)f5.x; h2[5]=(__bf16)f5.y; h2[6]=(__bf16)f5.z; h2[7]=(__bf16)f5.w;
          h3[0]=(__bf16)f6.x; h3[1]=(__bf16)f6.y; h3[2]=(__bf16)f6.z; h3[3]=(__bf16)f6.w;
          h3[4]=(__bf16)f7.x; h3[5]=(__bf16)f7.y; h3[6]=(__bf16)f7.z; h3[7]=(__bf16)f7.w;
        }
        *(bf16x8*)(&Bs[tid][0])  = h0;
        *(bf16x8*)(&Bs[tid][8])  = h1;
        *(bf16x8*)(&Bs[tid][16]) = h2;
        *(bf16x8*)(&Bs[tid][24]) = h3;
      } else {
        for (int i = tid; i < 128 * 32; i += 128) {
          int n = i >> 5, c = i & 31;
          int gn = n0 + n, gc = k0 + c;
          float v = 0.0f;
          if (gn < N && gc < K) v = B[offB + (long long)gn * ldb + gc];
          Bs[n][c] = (__bf16)v;
        }
      }
    } else {
      if (fullK) {
        // transpose through registers: K-pairs packed, b32 LDS stores
        #pragma unroll
        for (int it = 0; it < 16; ++it) {
          int p = tid + it * 128;
          int n = p & 127, kk2 = p >> 7;             // kk2: 0..15 -> k pair
          int gn = n0 + n;
          bf16x2 pr = {};
          if (gn < N) {
            const float* gp = B + offB + (long long)(k0 + 2 * kk2) * ldb + gn;
            pr[0] = (__bf16)gp[0];
            pr[1] = (__bf16)gp[ldb];
          }
          *(bf16x2*)(&Bs[n][2 * kk2]) = pr;
        }
        if (k0 + 32 < K)
          __builtin_prefetch(B + offB + (long long)(k0 + 32 + (tid >> 7) * 16) * ldb + n0 + (tid & 127), 0, 3);
      } else {
        for (int i = tid; i < 128 * 32; i += 128) {
          int kk = i >> 7, n = i & 127;
          int gk = k0 + kk, gn = n0 + n;
          float v = 0.0f;
          if (gk < K && gn < N) v = B[offB + (long long)gk * ldb + gn];
          Bs[n][kk] = (__bf16)v;
        }
      }
    }
    __syncthreads();

    // ---- build fragments (CDNA5 16-bit A/B striping) and issue 2x4 WMMA
    bf16x16 af[2], bfm[4];
    #pragma unroll
    for (int mi = 0; mi < 2; ++mi) {
      const __bf16* ap = &As[wm * 32 + mi * 16 + rsel][0];
      bf16x8 lo = *(const bf16x8*)(ap + 8 * khalf);        // K = 8h .. 8h+7
      bf16x8 hi = *(const bf16x8*)(ap + 16 + 8 * khalf);   // K = 16+8h .. +7
      af[mi] = __builtin_shufflevector(lo, hi, 0,1,2,3,4,5,6,7,8,9,10,11,12,13,14,15);
    }
    #pragma unroll
    for (int ni = 0; ni < 4; ++ni) {
      const __bf16* bp = &Bs[wn * 64 + ni * 16 + rsel][0];
      bf16x8 lo = *(const bf16x8*)(bp + 8 * khalf);        // K = 8h .. 8h+7
      bf16x8 hi = *(const bf16x8*)(bp + 16 + 8 * khalf);   // K = 16+8h .. +7
      bfm[ni] = __builtin_shufflevector(lo, hi, 0,1,2,3,4,5,6,7,8,9,10,11,12,13,14,15);
    }
    #pragma unroll
    for (int mi = 0; mi < 2; ++mi)
      #pragma unroll
      for (int ni = 0; ni < 4; ++ni)
        acc[mi][ni] = __builtin_amdgcn_wmma_f32_16x16x32_bf16(
            false, af[mi], false, bfm[ni], (short)0, acc[mi][ni], false, false);
    __syncthreads();
  }

  // ---- epilogue + store (C layout: lanes 0-15 N=lane M=i; 16-31 N=lane-16 M=i+8)
  #pragma unroll
  for (int mi = 0; mi < 2; ++mi)
    #pragma unroll
    for (int ni = 0; ni < 4; ++ni) {
      int col = n0 + wn * 64 + ni * 16 + rsel;
      float bv = (bias && col < N) ? bias[col] : 0.0f;
      #pragma unroll
      for (int i = 0; i < 8; ++i) {
        int row = m0 + wm * 32 + mi * 16 + i + 8 * khalf;
        if (row < M && col < N) {
          float v = acc[mi][ni][i] + bv;
          if (epi == 1)      v = fmaxf(v, 0.0f);
          else if (epi == 2) v = tanhf(v);
          else if (epi == 3) {
            int len = mlen[zo];
            v = (row < len && col < len) ? v * scale : NEGV;
          }
          C[offC + (long long)row * ldc + col] = v;
        }
      }
    }
}

// ---------------------------------------------------------------------------
// fc_in_gps + transpose:  x[b,l,h] = src[l,b,:3] @ W + bias
// ---------------------------------------------------------------------------
__global__ void k_fc_gps(const float* __restrict__ src, const float* __restrict__ W,
                         const float* __restrict__ b, float* __restrict__ x)
{
  int idx = blockIdx.x * blockDim.x + threadIdx.x;
  if (idx >= BS * LSQ * HID) return;
  int h = idx & (HID - 1);
  int bl = idx >> 9;
  int l = bl & (LSQ - 1), bb = bl >> 8;
  const float* s = src + ((long long)l * BS + bb) * 3;
  x[idx] = s[0] * W[h] + s[1] * W[HID + h] + s[2] * W[2 * HID + h] + b[h];
}

// ---------------------------------------------------------------------------
// Transpose [rows, cols] -> [cols, rows]   (for gru_wih hoist)
// ---------------------------------------------------------------------------
__global__ void k_transT(const float* __restrict__ srcM, float* __restrict__ dst,
                         int rows, int cols)
{
  int idx = blockIdx.x * blockDim.x + threadIdx.x;
  if (idx >= rows * cols) return;
  int r = idx / cols, c = idx % cols;
  dst[(long long)c * rows + r] = srcM[idx];
}

// ---------------------------------------------------------------------------
// Row softmax (one block per row, 256 threads)
// ---------------------------------------------------------------------------
__global__ __launch_bounds__(256)
void k_softmax(float* __restrict__ data, int width)
{
  __shared__ float red[256];
  long long row = blockIdx.x;
  float* p = data + row * width;
  int t = threadIdx.x;
  float mx = -3.4e38f;
  for (int c = t; c < width; c += 256) mx = fmaxf(mx, p[c]);
  red[t] = mx; __syncthreads();
  for (int s = 128; s > 0; s >>= 1) { if (t < s) red[t] = fmaxf(red[t], red[t + s]); __syncthreads(); }
  mx = red[0]; __syncthreads();
  float sm = 0.0f;
  for (int c = t; c < width; c += 256) { float e = __expf(p[c] - mx); p[c] = e; sm += e; }
  red[t] = sm; __syncthreads();
  for (int s = 128; s > 0; s >>= 1) { if (t < s) red[t] += red[t + s]; __syncthreads(); }
  float inv = 1.0f / red[0];
  for (int c = t; c < width; c += 256) p[c] *= inv;
}

// ---------------------------------------------------------------------------
// x = LayerNorm(x + t) * g + b   (rows of 512, one block per row)
// ---------------------------------------------------------------------------
__global__ __launch_bounds__(256)
void k_add_ln(float* __restrict__ x, const float* __restrict__ tt,
              const float* __restrict__ g, const float* __restrict__ b)
{
  __shared__ float red[256];
  __shared__ float row[HID];
  long long r = blockIdx.x;
  int t = threadIdx.x;
  float s = 0.0f;
  for (int c = t; c < HID; c += 256) { float v = x[r * HID + c] + tt[r * HID + c]; row[c] = v; s += v; }
  red[t] = s; __syncthreads();
  for (int st = 128; st > 0; st >>= 1) { if (t < st) red[t] += red[t + st]; __syncthreads(); }
  float mean = red[0] / (float)HID; __syncthreads();
  s = 0.0f;
  for (int c = t; c < HID; c += 256) { float d = row[c] - mean; s += d * d; }
  red[t] = s; __syncthreads();
  for (int st = 128; st > 0; st >>= 1) { if (t < st) red[t] += red[t + st]; __syncthreads(); }
  float rstd = rsqrtf(red[0] / (float)HID + 1e-5f); __syncthreads();
  for (int c = t; c < HID; c += 256)
    x[r * HID + c] = (row[c] - mean) * rstd * g[c] + b[c];
}

// ---------------------------------------------------------------------------
// Fused candidate attention per (b,l):
//   candidates built in LDS (emb gather + 9-dim matmul), score softmax, ctx,
//   fusion = LN(gps_bs + ctx), mm_probs (sigmoid masked) -> d_out
// ---------------------------------------------------------------------------
__global__ __launch_bounds__(256)
void k_candi(const float* __restrict__ x, const int* __restrict__ src_len,
             const int* __restrict__ cids, const float* __restrict__ cfeats,
             const int* __restrict__ cmask, const float* __restrict__ emb,
             const float* __restrict__ cw, const float* __restrict__ cb,
             const float* __restrict__ fg, const float* __restrict__ fb,
             float* __restrict__ fusion, float* __restrict__ mmout)
{
  __shared__ float C[KC][HID];
  __shared__ float g[HID];
  __shared__ float fl[HID];
  __shared__ float feats[KC][9];
  __shared__ float red[256];
  __shared__ float scs[KC];
  __shared__ int   ids[KC];
  int bl = blockIdx.x;
  int l = bl & (LSQ - 1), b = bl >> 8;
  int t = threadIdx.x;
  const long long base = (long long)bl * KC;
  if (t < KC) ids[t] = cids[base + t];
  if (t < KC * 9) feats[t / 9][t % 9] = cfeats[base * 9 + t];
  float vmul = (l < src_len[b]) ? 1.0f : 0.0f;
  for (int c = t; c < HID; c += 256) g[c] = x[(long long)bl * HID + c] * vmul;
  __syncthreads();
  for (int i = t; i < KC * HID; i += 256) {
    int k = i >> 9, h = i & (HID - 1);
    float v = emb[(long long)ids[k] * HID + h] + cb[h];
    for (int f = 0; f < 9; ++f) v += feats[k][f] * cw[f * HID + h];
    C[k][h] = v;
  }
  __syncthreads();
  for (int k = 0; k < KC; ++k) {
    float s = 0.0f;
    for (int c = t; c < HID; c += 256) s += g[c] * C[k][c];
    red[t] = s; __syncthreads();
    for (int st = 128; st > 0; st >>= 1) { if (t < st) red[t] += red[t + st]; __syncthreads(); }
    if (t == 0) scs[k] = (cmask[base + k] == 0) ? NEGV : red[0] * 0.04419417382f;
    __syncthreads();
  }
  float mx = -3.4e38f;
  for (int k = 0; k < KC; ++k) mx = fmaxf(mx, scs[k]);
  float aw[KC]; float sum = 0.0f;
  for (int k = 0; k < KC; ++k) { aw[k] = __expf(scs[k] - mx); sum += aw[k]; }
  float inv = 1.0f / sum;
  for (int k = 0; k < KC; ++k) aw[k] *= inv;
  for (int c = t; c < HID; c += 256) {
    float v = g[c];
    for (int k = 0; k < KC; ++k) v += aw[k] * C[k][c];
    fl[c] = v;
  }
  __syncthreads();
  float s = 0.0f;
  for (int c = t; c < HID; c += 256) s += fl[c];
  red[t] = s; __syncthreads();
  for (int st = 128; st > 0; st >>= 1) { if (t < st) red[t] += red[t + st]; __syncthreads(); }
  float mean = red[0] / (float)HID; __syncthreads();
  s = 0.0f;
  for (int c = t; c < HID; c += 256) { float d = fl[c] - mean; s += d * d; }
  red[t] = s; __syncthreads();
  for (int st = 128; st > 0; st >>= 1) { if (t < st) red[t] += red[t + st]; __syncthreads(); }
  float rstd = rsqrtf(red[0] / (float)HID + 1e-5f); __syncthreads();
  for (int c = t; c < HID; c += 256) {
    float v = (fl[c] - mean) * rstd * fg[c] + fb[c];
    fl[c] = v;
    fusion[(long long)bl * HID + c] = v;
  }
  __syncthreads();
  for (int k = 0; k < KC; ++k) {
    float s2 = 0.0f;
    for (int c = t; c < HID; c += 256) s2 += fl[c] * C[k][c];
    red[t] = s2; __syncthreads();
    for (int st = 128; st > 0; st >>= 1) { if (t < st) red[t] += red[t + st]; __syncthreads(); }
    if (t == 0)
      mmout[base + k] = (cmask[base + k] == 0) ? 0.0f : 1.0f / (1.0f + __expf(-red[0]));
    __syncthreads();
  }
}

// ---------------------------------------------------------------------------
// hcat[b,0:64] = temporal[pro[b]] ; hcat[b,64+h] = mean_l fusion (masked)
// ---------------------------------------------------------------------------
__global__ void k_hidden(const float* __restrict__ fusion, const int* __restrict__ src_len,
                         const int* __restrict__ pro, const float* __restrict__ temporal,
                         float* __restrict__ hcat)
{
  int idx = blockIdx.x * blockDim.x + threadIdx.x;
  if (idx >= BS * (POUT + HID)) return;
  int b = idx / (POUT + HID), j = idx % (POUT + HID);
  if (j < POUT) {
    hcat[idx] = temporal[(long long)pro[b] * POUT + j];
  } else {
    int h = j - POUT;
    int len = src_len[b];
    float s = 0.0f;
    for (int l = 0; l < len; ++l) s += fusion[((long long)b * LSQ + l) * HID + h];
    hcat[idx] = s / (float)len;
  }
}

// ---------------------------------------------------------------------------
// Decoder precompute gathers
// ---------------------------------------------------------------------------
__global__ void k_pemb(const int* __restrict__ path, const float* __restrict__ emb,
                       float* __restrict__ pe)
{
  int idx = blockIdx.x * blockDim.x + threadIdx.x;
  if (idx >= BS * PPN * HID) return;
  int h = idx & (HID - 1);
  int bp = idx >> 9;
  int p = bp & (PPN - 1), b = bp >> 7;
  pe[idx] = emb[(long long)path[p * BS + b] * HID + h];
}

__global__ void k_dextra(const int* __restrict__ d_rids, const float* __restrict__ emb,
                         const float* __restrict__ ridf, const float* __restrict__ d_rates,
                         float* __restrict__ de)
{
  int idx = blockIdx.x * blockDim.x + threadIdx.x;
  if (idx >= BS * (HID + RFD + 1)) return;
  int b = idx / (HID + RFD + 1), j = idx % (HID + RFD + 1);
  int rid = d_rids[b];
  float v;
  if (j < HID)            v = emb[(long long)rid * HID + j];
  else if (j < HID + RFD) v = ridf[(long long)rid * RFD + (j - HID)];
  else                    v = d_rates[b];
  de[idx] = v;
}

// x_in written with padded row stride XPAD (cols >= INDIM never read)
__global__ void k_xin(const int* __restrict__ tid_t, const float* __restrict__ trate_t,
                      const float* __restrict__ emb, const float* __restrict__ ridf,
                      const float* __restrict__ de, float* __restrict__ xin)
{
  int idx = blockIdx.x * blockDim.x + threadIdx.x;
  if (idx >= BS * INDIM) return;
  int b = idx / INDIM, j = idx % INDIM;
  float v;
  if (j < HID)                 v = emb[(long long)tid_t[b] * HID + j];
  else if (j < HID + RFD)      v = ridf[(long long)tid_t[b] * RFD + (j - HID)];
  else if (j == HID + RFD)     v = trate_t[b];
  else                         v = de[b * (HID + RFD + 1) + (j - (HID + RFD + 1))];
  xin[(long long)b * XPAD + j] = v;
}

// ---------------------------------------------------------------------------
// GRU elementwise update (in place on h)
// ---------------------------------------------------------------------------
__global__ void k_gru(const float* __restrict__ gi, const float* __restrict__ gh,
                      float* __restrict__ h)
{
  int idx = blockIdx.x * blockDim.x + threadIdx.x;
  if (idx >= BS * HID) return;
  int b = idx >> 9, c = idx & (HID - 1);
  const float* pi = gi + (long long)b * 3 * HID;
  const float* ph = gh + (long long)b * 3 * HID;
  float r = 1.0f / (1.0f + __expf(-(pi[c] + ph[c])));
  float z = 1.0f / (1.0f + __expf(-(pi[HID + c] + ph[HID + c])));
  float n = tanhf(pi[2 * HID + c] + r * ph[2 * HID + c]);
  h[idx] = (1.0f - z) * n + z * h[idx];
}

// ---------------------------------------------------------------------------
// Fused decoder attention per batch b (128 threads = one per path position):
//   sc2 = tanh(pe_proj + h_proj) . attn_v  -> masked softmax -> pid (to d_out)
//   wp = pid @ path_emb ; ri = [h2, wp]
// ---------------------------------------------------------------------------
__global__ __launch_bounds__(128)
void k_dec_attn(const float* __restrict__ h, const float* __restrict__ hproj,
                const float* __restrict__ peproj, const float* __restrict__ pemb,
                const float* __restrict__ attn_v, const int* __restrict__ path_len,
                float* __restrict__ ids_out, float* __restrict__ ri)
{
  __shared__ float hp[HID];
  __shared__ float av[HID];
  __shared__ float pid[PPN];
  __shared__ float red[PPN];
  int b = blockIdx.x;
  int t = threadIdx.x;
  for (int c = t; c < HID; c += 128) { hp[c] = hproj[(long long)b * HID + c]; av[c] = attn_v[c]; }
  __syncthreads();
  const float* per = peproj + ((long long)b * PPN + t) * HID;
  float s = 0.0f;
  for (int c = 0; c < HID; ++c) s += tanhf(per[c] + hp[c]) * av[c];
  s = (t < path_len[b]) ? s : NEGV;
  red[t] = s; __syncthreads();
  for (int st = 64; st > 0; st >>= 1) { if (t < st) red[t] = fmaxf(red[t], red[t + st]); __syncthreads(); }
  float mx = red[0]; __syncthreads();
  float e = __expf(s - mx);
  red[t] = e; __syncthreads();
  for (int st = 64; st > 0; st >>= 1) { if (t < st) red[t] += red[t + st]; __syncthreads(); }
  float p = e / red[0];
  pid[t] = p;
  ids_out[(long long)b * PPN + t] = p;
  __syncthreads();
  for (int hh = t; hh < HID; hh += 128) {
    float w = 0.0f;
    for (int pp = 0; pp < PPN; ++pp) w += pid[pp] * pemb[((long long)b * PPN + pp) * HID + hh];
    ri[(long long)b * 2 * HID + HID + hh] = w;
    ri[(long long)b * 2 * HID + hh]       = h[(long long)b * HID + hh];
  }
}

// ---------------------------------------------------------------------------
// rate head stage 2: pr = sigmoid(r1 . rate_w2 + b2)
// ---------------------------------------------------------------------------
__global__ __launch_bounds__(128)
void k_rate2(const float* __restrict__ r1, const float* __restrict__ w2,
             const float* __restrict__ b2, float* __restrict__ out)
{
  __shared__ float red[128];
  int b = blockIdx.x, t = threadIdx.x;
  float s = 0.0f;
  for (int c = t; c < 2 * HID; c += 128) s += r1[(long long)b * 2 * HID + c] * w2[c];
  red[t] = s; __syncthreads();
  for (int st = 64; st > 0; st >>= 1) { if (t < st) red[t] += red[t + st]; __syncthreads(); }
  if (t == 0) out[b] = 1.0f / (1.0f + __expf(-(red[0] + b2[0])));
}

// ---------------------------------------------------------------------------
// Host-side orchestration
// ---------------------------------------------------------------------------
extern "C" void kernel_launch(void* const* d_in, const int* in_sizes, int n_in,
                              void* d_out, int out_size, void* d_ws, size_t ws_size,
                              hipStream_t stream)
{
  (void)in_sizes; (void)n_in; (void)out_size; (void)ws_size;
  // ---- inputs (setup_inputs dict order; params dict flattened in order)
  const float* src        = (const float*)d_in[0];
  const int*   src_len    = (const int*)  d_in[1];
  const int*   candi_ids  = (const int*)  d_in[2];
  const float* candi_fts  = (const float*)d_in[3];
  const int*   candi_msk  = (const int*)  d_in[4];
  const int*   trg_id     = (const int*)  d_in[5];
  const float* trg_rate   = (const float*)d_in[6];
  const int*   pro        = (const int*)  d_in[7];
  const float* ridf       = (const float*)d_in[8];
  const int*   path       = (const int*)  d_in[9];
  const int*   path_len   = (const int*)  d_in[10];
  const int*   d_rids     = (const int*)  d_in[11];
  const float* d_rates    = (const float*)d_in[12];
  const float* emb        = (const float*)d_in[13];
  const float* fc_gps_w   = (const float*)d_in[14];
  const float* fc_gps_b   = (const float*)d_in[15];
  const float* Wq = (const float*)d_in[16]; const float* bq = (const float*)d_in[17];
  const float* Wk = (const float*)d_in[18]; const float* bk = (const float*)d_in[19];
  const float* Wv = (const float*)d_in[20]; const float* bv = (const float*)d_in[21];
  const float* Wo = (const float*)d_in[22]; const float* bo = (const float*)d_in[23];
  const float* W1 = (const float*)d_in[24]; const float* b1 = (const float*)d_in[25];
  const float* W2 = (const float*)d_in[26]; const float* b2 = (const float*)d_in[27];
  const float* ln1g = (const float*)d_in[28]; const float* ln1b = (const float*)d_in[29];
  const float* ln2g = (const float*)d_in[30]; const float* ln2b = (const float*)d_in[31];
  const float* candi_w = (const float*)d_in[32]; const float* candi_b = (const float*)d_in[33];
  const float* fn_g = (const float*)d_in[34]; const float* fn_b = (const float*)d_in[35];
  const float* temporal = (const float*)d_in[36];
  const float* fc_hid_w = (const float*)d_in[37]; const float* fc_hid_b = (const float*)d_in[38];
  const float* gru_wih = (const float*)d_in[39]; const float* gru_whh = (const float*)d_in[40];
  const float* gru_bih = (const float*)d_in[41]; const float* gru_bhh = (const float*)d_in[42];
  const float* attn_w = (const float*)d_in[43]; const float* attn_wb = (const float*)d_in[44];
  const float* attn_v = (const float*)d_in[45];
  const float* rate_w1 = (const float*)d_in[46]; const float* rate_b1 = (const float*)d_in[47];
  const float* rate_w2 = (const float*)d_in[48]; const float* rate_b2 = (const float*)d_in[49];

  // ---- outputs
  float* out      = (float*)d_out;
  float* mm_out   = out;                               // [32,256,10]
  float* ids_out  = out + BS * LSQ * KC;               // [62,32,128]
  float* rate_out = ids_out + (long long)(TTK - 2) * BS * PPN; // [62,32,1]

  // ---- workspace carve-up
  float* w = (float*)d_ws;
  size_t cur = 0;
  auto alloc = [&](size_t n) { float* p = w + cur; cur += n; return p; };
  const size_t SZA = (size_t)BS * LSQ * HID;           // 4,194,304
  float* xbuf   = alloc(SZA);
  float* qbuf   = alloc(SZA);
  float* kbuf   = alloc(SZA);
  float* vbuf   = alloc(SZA);
  float* ctxbuf = alloc(SZA);
  float* tmpbuf = alloc(SZA);
  float* fusion = alloc(SZA);
  float* big    = alloc((size_t)BS * LSQ * FFN);       // scores / FFN alias
  float* pemb   = alloc((size_t)BS * PPN * HID);
  float* peproj = alloc((size_t)BS * PPN * HID);
  float* wihT   = alloc((size_t)INDIM * 3 * HID);      // gru_wih transposed
  float* hbuf   = alloc((size_t)BS * HID);
  float* hcat   = alloc((size_t)BS * (POUT + HID));
  float* dex    = alloc((size_t)BS * (HID + RFD + 1));
  float* xin    = alloc((size_t)BS * XPAD);
  float* gi     = alloc((size_t)BS * 3 * HID);
  float* gh     = alloc((size_t)BS * 3 * HID);
  float* hproj  = alloc((size_t)BS * HID);
  float* ri     = alloc((size_t)BS * 2 * HID);
  float* r1     = alloc((size_t)BS * 2 * HID);

  auto gemm = [&](const float* A, const float* B, const float* bias, float* C,
                  int M, int N, int K, int lda, int ldb, int ldc, bool transB,
                  int batches, int heads,
                  long long sAo, long long sAi, long long sBo, long long sBi,
                  long long sCo, long long sCi,
                  int epi, float scale, const int* mlen) {
    dim3 g((N + 127) / 128, (M + 63) / 64, batches);
    dim3 blk(128);
    if (transB)
      gemm_wmma<true><<<g, blk, 0, stream>>>(A, B, bias, C, M, N, K, lda, ldb, ldc,
                                             heads, sAo, sAi, sBo, sBi, sCo, sCi, epi, scale, mlen);
    else
      gemm_wmma<false><<<g, blk, 0, stream>>>(A, B, bias, C, M, N, K, lda, ldb, ldc,
                                              heads, sAo, sAi, sBo, sBi, sCo, sCi, epi, scale, mlen);
  };

  const int MT = BS * LSQ;   // 8192 tokens

  // ================= encoder input projection =================
  k_fc_gps<<<(BS * LSQ * HID + 255) / 256, 256, 0, stream>>>(src, fc_gps_w, fc_gps_b, xbuf);

  // ================= transformer encoder =================
  for (int ly = 0; ly < NLAY; ++ly) {
    const float* wq = Wq + (long long)ly * HID * HID;
    const float* wk = Wk + (long long)ly * HID * HID;
    const float* wv = Wv + (long long)ly * HID * HID;
    const float* wo = Wo + (long long)ly * HID * HID;
    gemm(xbuf, wq, bq + ly * HID, qbuf, MT, HID, HID, HID, HID, HID, false,
         1, 1, 0, 0, 0, 0, 0, 0, 0, 1.0f, nullptr);
    gemm(xbuf, wk, bk + ly * HID, kbuf, MT, HID, HID, HID, HID, HID, false,
         1, 1, 0, 0, 0, 0, 0, 0, 0, 1.0f, nullptr);
    gemm(xbuf, wv, bv + ly * HID, vbuf, MT, HID, HID, HID, HID, HID, false,
         1, 1, 0, 0, 0, 0, 0, 0, 0, 1.0f, nullptr);
    // scores s[b,h,i,j] = q . k / 8, masked -> big
    gemm(qbuf, kbuf, nullptr, big, LSQ, LSQ, DH, HID, HID, LSQ, true,
         BS * HEADS, HEADS,
         (long long)LSQ * HID, DH,
         (long long)LSQ * HID, DH,
         (long long)HEADS * LSQ * LSQ, (long long)LSQ * LSQ,
         3, 0.125f, src_len);
    k_softmax<<<BS * HEADS * LSQ, 256, 0, stream>>>(big, LSQ);
    // ctx[b,i,h*64+d] = a @ v
    gemm(big, vbuf, nullptr, ctxbuf, LSQ, DH, LSQ, LSQ, HID, HID, false,
         BS * HEADS, HEADS,
         (long long)HEADS * LSQ * LSQ, (long long)LSQ * LSQ,
         (long long)LSQ * HID, DH,
         (long long)LSQ * HID, DH,
         0, 1.0f, nullptr);
    gemm(ctxbuf, wo, bo + ly * HID, tmpbuf, MT, HID, HID, HID, HID, HID, false,
         1, 1, 0, 0, 0, 0, 0, 0, 0, 1.0f, nullptr);
    k_add_ln<<<MT, 256, 0, stream>>>(xbuf, tmpbuf, ln1g + ly * HID, ln1b + ly * HID);
    gemm(xbuf, W1 + (long long)ly * HID * FFN, b1 + ly * FFN, big, MT, FFN, HID,
         HID, FFN, FFN, false, 1, 1, 0, 0, 0, 0, 0, 0, 1, 1.0f, nullptr);
    gemm(big, W2 + (long long)ly * FFN * HID, b2 + ly * HID, tmpbuf, MT, HID, FFN,
         FFN, HID, HID, false, 1, 1, 0, 0, 0, 0, 0, 0, 0, 1.0f, nullptr);
    k_add_ln<<<MT, 256, 0, stream>>>(xbuf, tmpbuf, ln2g + ly * HID, ln2b + ly * HID);
  }

  // ================= candidate attention + fusion + mm_probs =================
  k_candi<<<MT, 256, 0, stream>>>(xbuf, src_len, candi_ids, candi_fts, candi_msk,
                                  emb, candi_w, candi_b, fn_g, fn_b, fusion, mm_out);

  // ================= hidden init =================
  k_hidden<<<(BS * (POUT + HID) + 255) / 256, 256, 0, stream>>>(fusion, src_len, pro, temporal, hcat);
  gemm(hcat, fc_hid_w, fc_hid_b, hbuf, BS, HID, POUT + HID, POUT + HID, HID, HID, false,
       1, 1, 0, 0, 0, 0, 0, 0, 2 /*tanh*/, 1.0f, nullptr);

  // ================= decoder precompute =================
  k_pemb<<<(BS * PPN * HID + 255) / 256, 256, 0, stream>>>(path, emb, pemb);
  // pe_proj = path_emb @ attn_w[512:,:] + attn_wb   (step-invariant hoist)
  gemm(pemb, attn_w + (long long)HID * HID, attn_wb, peproj, BS * PPN, HID, HID,
       HID, HID, HID, false, 1, 1, 0, 0, 0, 0, 0, 0, 0, 1.0f, nullptr);
  k_dextra<<<(BS * (HID + RFD + 1) + 255) / 256, 256, 0, stream>>>(d_rids, emb, ridf, d_rates, dex);
  // gru_wih [3H, INDIM] -> wihT [INDIM, 3H]  (enables non-trans B fast path)
  k_transT<<<(3 * HID * INDIM + 255) / 256, 256, 0, stream>>>(gru_wih, wihT, 3 * HID, INDIM);

  // ================= decoder: 62 effective steps =================
  for (int t = 0; t < TTK - 2; ++t) {
    k_xin<<<(BS * INDIM + 255) / 256, 256, 0, stream>>>(trg_id + t * BS, trg_rate + t * BS,
                                                        emb, ridf, dex, xin);
    gemm(xin, wihT, gru_bih, gi, BS, 3 * HID, INDIM, XPAD, 3 * HID, 3 * HID, false,
         1, 1, 0, 0, 0, 0, 0, 0, 0, 1.0f, nullptr);
    gemm(hbuf, gru_whh, gru_bhh, gh, BS, 3 * HID, HID, HID, HID, 3 * HID, true,
         1, 1, 0, 0, 0, 0, 0, 0, 0, 1.0f, nullptr);
    k_gru<<<(BS * HID + 255) / 256, 256, 0, stream>>>(gi, gh, hbuf);
    gemm(hbuf, attn_w, nullptr, hproj, BS, HID, HID, HID, HID, HID, false,
         1, 1, 0, 0, 0, 0, 0, 0, 0, 1.0f, nullptr);
    k_dec_attn<<<BS, 128, 0, stream>>>(hbuf, hproj, peproj, pemb, attn_v, path_len,
                                       ids_out + (long long)t * BS * PPN, ri);
    gemm(ri, rate_w1, rate_b1, r1, BS, 2 * HID, 2 * HID, 2 * HID, 2 * HID, 2 * HID, false,
         1, 1, 0, 0, 0, 0, 0, 0, 1 /*relu*/, 1.0f, nullptr);
    k_rate2<<<BS, 128, 0, stream>>>(r1, rate_w2, rate_b2, rate_out + t * BS);
  }
}
